// SoftLabelPropagation_59416577573254
// MI455X (gfx1250) — matrix operand order repaired and checked by reading, not compiled, not run on hardware
//
#include <hip/hip_runtime.h>
#include <hip/hip_bf16.h>
#include <math.h>

// ---------------- problem constants (from reference) ----------------
#define RUNS   500
#define WAYS   5
#define SHOT   5
#define NS     25        // support rows
#define QS     75        // query rows
#define NN     100       // total rows
#define NP     112       // padded rows (7 x 16 WMMA tiles)
#define NPK    128       // K padding for bf16 GEMMs over N
#define DIM    640
#define LAMBDA 10.0f
#define ALPHA  0.7f
#define UPD    0.6f
#define KHOP   4
#define NEPOCH 10
#define LINKK  6
#define SEPS   1e-3f
#define MAXIT  1000
#define BLOCK  256

typedef __bf16 bf16;
typedef __attribute__((ext_vector_type(16))) __bf16 v16bf;
typedef __attribute__((ext_vector_type(8)))  float  v8f;
typedef __attribute__((ext_vector_type(2)))  float  v2f;

// ---------------- LDS arena ----------------
#define OFF_M0 0
#define OFF_M1 (NP*NP*4)                 //  50176
#define OFF_GB (2*NP*NP*4)               // 100352
#define OFF_FC (OFF_GB + NP*NPK*2)       // 129024
#define OFF_SM (OFF_FC + NPK*16*2)       // 133120
#define SMEM_BYTES 135232

__device__ __forceinline__ int laneid() { return threadIdx.x & 31; }

__device__ __forceinline__ v8f vzero8() {
  v8f c; 
#pragma unroll
  for (int e = 0; e < 8; ++e) c[e] = 0.f;
  return c;
}

// ---- bf16 WMMA fragments (this code's contiguous k order k = 16*(lane/16) + e, not the ISA table order) ----
// A 16x32: lanes0-15 row=lane, K={0..15}; lanes16-31 same rows, K={16..31}
__device__ __forceinline__ v16bf frag_a_bf16(const bf16* base, int ld, int row0, int k0) {
  int ln = laneid();
  const bf16* p = base + (row0 + (ln & 15)) * ld + k0 + ((ln >> 4) << 4);
  v16bf a;
#pragma unroll
  for (int e = 0; e < 8; ++e) { a[e] = p[e]; a[e + 8] = p[e + 8]; }
  return a;
}
// B 32x16 where B[k][n] = base[(row0+n)*ld + k]  (B = rowsᵀ, contiguous in k)
__device__ __forceinline__ v16bf frag_b_bf16_rowsT(const bf16* base, int ld, int row0, int k0) {
  int ln = laneid();
  const bf16* p = base + (row0 + (ln & 15)) * ld + k0 + ((ln >> 4) << 4);
  return *(const v16bf*)p;   // 16 consecutive bf16, 32B aligned
}
// B 32x16 where B[k][n] = base[k*ld + col0+n]  (k-major source)
__device__ __forceinline__ v16bf frag_b_bf16_km(const bf16* base, int ld, int k0, int col0) {
  int ln = laneid();
  int kb = k0 + ((ln >> 4) << 4);
  int c  = col0 + (ln & 15);
  v16bf b;
#pragma unroll
  for (int e = 0; e < 16; ++e) b[e] = base[(kb + e) * ld + c];
  return b;
}
// A = Z2ᵀ on the fly (f32 LDS -> bf16), rows=ways, K=i (guard k<NP)
__device__ __forceinline__ v16bf frag_a_zT(const float* Zf, int k0) {
  int ln = laneid(); int w = ln & 15; int kb = k0 + ((ln >> 4) << 4);
  v16bf a;
#pragma unroll
  for (int e = 0; e < 8; ++e) {
    int k1 = kb + e, k2 = kb + 8 + e;
    a[e]     = (k1 < NP) ? (bf16)Zf[k1 * 16 + w] : (bf16)0.f;
    a[e + 8] = (k2 < NP) ? (bf16)Zf[k2 * 16 + w] : (bf16)0.f;
  }
  return a;
}
// B from global features with K guard (k<NP)
__device__ __forceinline__ v16bf frag_b_F_guard(const bf16* F, int k0, int col0) {
  int ln = laneid(); int kb = k0 + ((ln >> 4) << 4); int c = col0 + (ln & 15);
  v16bf b;
#pragma unroll
  for (int e = 0; e < 16; ++e) { int k = kb + e; b[e] = (k < NP) ? F[k * DIM + c] : (bf16)0.f; }
  return b;
}
// ---- f32 16x16x4 fragments: k = e + 2*(lane>=16) ----
__device__ __forceinline__ v2f frag_a_f32(const float* base, int ld, int row0, int k0) {
  int ln = laneid();
  const float* p = base + (row0 + (ln & 15)) * ld + k0 + ((ln >> 4) << 1);
  v2f a; a[0] = p[0]; a[1] = p[1]; return a;
}
__device__ __forceinline__ v2f frag_b_f32(const float* base, int ld, int k0, int col0) {
  int ln = laneid();
  int kb = k0 + ((ln >> 4) << 1); int c = col0 + (ln & 15);
  v2f b; b[0] = base[kb * ld + c]; b[1] = base[(kb + 1) * ld + c]; return b;
}

__device__ __forceinline__ v8f wmma_bf16(v16bf a, v16bf b, v8f c) {
  return __builtin_amdgcn_wmma_f32_16x16x32_bf16(false, a, false, b, (short)0, c, false, false);
}
__device__ __forceinline__ v8f wmma_f32(v2f a, v2f b, v8f c) {
  return __builtin_amdgcn_wmma_f32_16x16x4_f32(false, a, false, b, (short)0, c, false, false);
}

// ---------------- in-block Sinkhorn (matches reference cond/body order) ----------------
__device__ void sinkhorn(float* P, int r0, int r1, float cval, int nl,
                         const int* ysL, float* ust, float* rsA, int* flg) {
  int tid = threadIdx.x;
  if (tid >= r0 && tid < r1) ust[tid] = 0.f;
  __syncthreads();
  int it = 1;
  while (true) {
    if (tid >= r0 && tid < r1) {
      float s = 0.f;
#pragma unroll
      for (int w = 0; w < WAYS; ++w) s += P[tid * 16 + w];
      rsA[tid] = s;
    }
    __syncthreads();
    if (tid == 0) {
      float md = 0.f;
      for (int i = r0; i < r1; ++i) md = fmaxf(md, fabsf(ust[i] - rsA[i]));
      flg[0] = (md > SEPS && it <= MAXIT) ? 1 : 0;
    }
    __syncthreads();
    if (!flg[0]) break;
    if (tid >= r0 && tid < r1) {
      float inv = 1.f / rsA[tid];
#pragma unroll
      for (int w = 0; w < WAYS; ++w) P[tid * 16 + w] *= inv;
      ust[tid] = rsA[tid];
    }
    __syncthreads();
    if (tid < WAYS) {
      float s = 0.f;
      for (int i = r0; i < r1; ++i) s += P[i * 16 + tid];
      float sc = cval / s;
      for (int i = r0; i < r1; ++i) P[i * 16 + tid] *= sc;
    }
    __syncthreads();
    if (nl > 0 && tid < nl) {
#pragma unroll
      for (int w = 0; w < WAYS; ++w) P[tid * 16 + w] = (ysL[tid] == w) ? 1.f : 0.f;
    }
    ++it;
    __syncthreads();
  }
}

// ---------------- get_prob: Z = [one_hot(ys) ; sinkhorn(exp(-lam*d2(Fq,proto)))] ----------------
__device__ void get_prob(const bf16* F, const float* protoF, bf16* protoT, float* Z,
                         const float* nrm, float* pn, const int* ysL,
                         float* ust, float* rsA, int* flg) {
  int tid = threadIdx.x, wid = tid >> 5;
  // build protoᵀ (bf16, [DIM][16]) and proto norms
  for (int i = tid; i < DIM * 16; i += BLOCK) {
    int d = i >> 4, w = i & 15;
    protoT[i] = (w < WAYS) ? (bf16)protoF[w * DIM + d] : (bf16)0.f;
  }
  if (tid < WAYS) {
    float s = 0.f;
    for (int d = 0; d < DIM; ++d) { float v = protoF[tid * DIM + d]; s += v * v; }
    pn[tid] = s;
  }
  __syncthreads();
  // dist GEMM: F [112x640] x protoᵀ [640x16] -> Z (with exp / one-hot / masking)
  if (wid < 7) {
    int ti = wid;
    v8f c = vzero8();
    for (int k0 = 0; k0 < DIM; k0 += 32)
      c = wmma_bf16(frag_a_bf16(F, DIM, ti * 16, k0), frag_b_bf16_km(protoT, 16, k0, 0), c);
    int ln = laneid(); int n = ln & 15;
#pragma unroll
    for (int e = 0; e < 8; ++e) {
      int m = e + ((ln >> 4) << 3); int i = ti * 16 + m;
      float v = 0.f;
      if (i < NN && n < WAYS) {
        if (i < NS) v = (ysL[i] == n) ? 1.f : 0.f;
        else {
          float d = fmaxf(nrm[i] + pn[n] - 2.f * c[e], 0.f);
          v = __expf(-LAMBDA * d);
        }
      }
      Z[i * 16 + n] = v;
    }
  }
  __syncthreads();
  sinkhorn(Z, NS, NN, (float)(QS / WAYS), 0, ysL, ust, rsA, flg);
}

// ---------------- main kernel: one workgroup (8 wave32) per episode ----------------
__global__ __launch_bounds__(BLOCK)
void slp_kernel(const float* __restrict__ xs, const float* __restrict__ xq,
                const int* __restrict__ ys, const int* __restrict__ yq,
                bf16* __restrict__ Fg, float* __restrict__ out) {
  extern __shared__ char smem[];
  float* M0  = (float*)(smem + OFF_M0);   // [NP][NP] adjacency / L / (I-aW)
  float* M1  = (float*)(smem + OFF_M1);   // [NP][NP] G / Inv
  bf16*  Gb  = (bf16*)(smem + OFF_GB);    // [NP][NPK] bf16 copy of G
  bf16*  Fc  = (bf16*)(smem + OFF_FC);    // [NPK][16] feature column chunk
  float* nrm = (float*)(smem + OFF_SM);   // [NP]
  float* dco = nrm + NP;                  // [NP]
  float* ust = dco + NP;                  // [NP]
  float* rsA = ust + NP;                  // [NP]
  float* pn  = rsA + NP;                  // [16]
  float* zs  = pn + 16;                   // [16]
  int*   ysL = (int*)(zs + 16);           // [32]
  int*   flg = ysL + 32;                  // [4]
  // phase-C overlays (M0 and Gb regions are free after the inverse is built)
  float* protoF = (float*)(smem + OFF_M0);             // [8][DIM] f32
  bf16*  protoT = (bf16*)(smem + OFF_M0 + 8*DIM*4);    // [DIM][16] bf16
  float* Z  = (float*)(smem + OFF_GB);                 // [NP][16]
  float* Z2 = Z + NP * 16;                             // [NP][16]

  const int tid = threadIdx.x, wid = tid >> 5, ln = tid & 31;
  const int b = blockIdx.x;
  bf16* F = Fg + (size_t)b * NP * DIM;

  // ---------- phase 0: features -> bf16 workspace (rows 100..111 zero) ----------
  {
    const float* sx = xs + (size_t)b * NS * DIM;
    const float* qx = xq + (size_t)b * QS * DIM;
    for (int i = tid; i < NP * DIM; i += BLOCK) {
      int r = i / DIM, d = i - r * DIM;
      float v = (r < NS) ? sx[r * DIM + d] : (r < NN ? qx[(r - NS) * DIM + d] : 0.f);
      F[i] = (bf16)v;
    }
    if (tid < NS) ysL[tid] = ys[b * NS + tid];
  }
  __syncthreads();

  // ---------- graph-conv hops ----------
  for (int hop = 0; hop < KHOP; ++hop) {
    // row norms
    if (tid < NP) {
      float s = 0.f;
      if (tid < NN) for (int d = 0; d < DIM; ++d) { float v = (float)F[tid * DIM + d]; s += v * v; }
      nrm[tid] = s;
    }
    __syncthreads();
    // A = exp(-lam * d2(F,F)) via bf16 WMMA Gram matrix (masked beyond NN)
    for (int t = wid; t < 49; t += 8) {
      int ti = t / 7, tj = t - ti * 7;
      v8f c = vzero8();
      for (int k0 = 0; k0 < DIM; k0 += 32)
        c = wmma_bf16(frag_a_bf16(F, DIM, ti * 16, k0),
                      frag_b_bf16_rowsT(F, DIM, tj * 16, k0), c);
      int n = tj * 16 + (ln & 15);
#pragma unroll
      for (int e = 0; e < 8; ++e) {
        int i = ti * 16 + e + ((ln >> 4) << 3);
        float d = fmaxf(nrm[i] + nrm[n] - 2.f * c[e], 0.f);
        M0[i * NP + n] = (i < NN && n < NN) ? __expf(-LAMBDA * d) : 0.f;
      }
    }
    __syncthreads();
    // top-k (k=6) per row, zero the rest
    if (tid < NN) {
      float* row = M0 + tid * NP;
      int idx[LINKK];
      for (int s = 0; s < LINKK; ++s) {
        float best = -1.f; int bi = 0;
        for (int j = 0; j < NN; ++j) {
          bool taken = false;
          for (int q = 0; q < s; ++q) taken |= (idx[q] == j);
          float v = row[j];
          if (!taken && v > best) { best = v; bi = j; }
        }
        idx[s] = bi;
      }
      for (int j = 0; j < NN; ++j) {
        bool keep = false;
        for (int q = 0; q < LINKK; ++q) keep |= (idx[q] == j);
        if (!keep) row[j] = 0.f;
      }
    }
    __syncthreads();
    // D^-1/2
    if (tid < NP) {
      float s = 0.f;
      for (int j = 0; j < NP; ++j) s += M0[tid * NP + j];
      dco[tid] = (s > 0.f) ? rsqrtf(s) : 0.f;
    }
    __syncthreads();
    // L = 0.5*I + 0.5*(D^-1/2 A D^-1/2)   (identity only on real rows)
    for (int i = tid; i < NP * NP; i += BLOCK) {
      int r = i / NP, cc = i - r * NP;
      float v = 0.5f * M0[i] * dco[r] * dco[cc];
      if (r == cc && r < NN) v += 0.5f;
      M0[i] = v;
    }
    __syncthreads();
    // G = L @ L  (exact f32 WMMA), also stash bf16 copy for G@F
    for (int t = wid; t < 49; t += 8) {
      int ti = t / 7, tj = t - ti * 7;
      v8f c = vzero8();
      for (int k0 = 0; k0 < NP; k0 += 4)
        c = wmma_f32(frag_a_f32(M0, NP, ti * 16, k0), frag_b_f32(M0, NP, k0, tj * 16), c);
      int n = tj * 16 + (ln & 15);
#pragma unroll
      for (int e = 0; e < 8; ++e) {
        int i = ti * 16 + e + ((ln >> 4) << 3);
        M1[i * NP + n] = c[e];
        Gb[i * NPK + n] = (bf16)c[e];
      }
    }
    for (int i = tid; i < NP * 16; i += BLOCK) {          // zero K-pad cols 112..127
      int r = i >> 4; Gb[r * NPK + NP + (i & 15)] = (bf16)0.f;
    }
    __syncthreads();
    // F <- G @ F, one 16-col chunk at a time (chunk staged in LDS so in-place is safe)
    for (int ch = 0; ch < DIM / 16; ++ch) {
      for (int i = tid; i < NPK * 16; i += BLOCK) {
        int r = i >> 4;
        Fc[i] = (r < NP) ? F[r * DIM + ch * 16 + (i & 15)] : (bf16)0.f;
      }
      __syncthreads();
      if (wid < 7) {
        int ti = wid;
        v8f c = vzero8();
        for (int k0 = 0; k0 < NPK; k0 += 32)
          c = wmma_bf16(frag_a_bf16(Gb, NPK, ti * 16, k0), frag_b_bf16_km(Fc, 16, k0, 0), c);
        int n = ln & 15;
#pragma unroll
        for (int e = 0; e < 8; ++e) {
          int i = ti * 16 + e + ((ln >> 4) << 3);
          F[i * DIM + ch * 16 + n] = (bf16)c[e];
        }
      }
      __syncthreads();
    }
  }

  // ---------- W = build_graph(build_graph(G)) on M1 ----------
  for (int rep = 0; rep < 2; ++rep) {
    if (tid < NP) M1[tid * NP + tid] = 0.f;
    __syncthreads();
    if (tid < NP) {
      float s = 0.f;
      for (int j = 0; j < NP; ++j) s += M1[tid * NP + j];
      dco[tid] = (s > 0.f) ? rsqrtf(s) : 0.f;
    }
    __syncthreads();
    for (int i = tid; i < NP * NP; i += BLOCK) {
      int r = i / NP, cc = i - r * NP;
      M1[i] = M1[i] * dco[r] * dco[cc];
    }
    __syncthreads();
  }

  // ---------- Inv = (I - alpha*W)^-1 by Gauss-Jordan (M0 -> reduced, M1 -> Inv) ----------
  for (int i = tid; i < NP * NP; i += BLOCK) {
    int r = i / NP, cc = i - r * NP;
    float w = M1[i];
    M0[i] = (r == cc ? 1.f : 0.f) - ALPHA * w;
    M1[i] = (r == cc) ? 1.f : 0.f;
  }
  __syncthreads();
  for (int p = 0; p < NP; ++p) {
    float pinv = 1.f / M0[p * NP + p];
    __syncthreads();
    for (int j = tid; j < NP; j += BLOCK) { M0[p * NP + j] *= pinv; M1[p * NP + j] *= pinv; }
    __syncthreads();
    int r = tid >> 1, half = tid & 1;
    float f = (tid < 2 * NP && r != p) ? M0[r * NP + p] : 0.f;
    __syncthreads();
    if (tid < 2 * NP && r != p) {
      int j0 = half * (NP / 2);
      for (int j = j0; j < j0 + NP / 2; ++j) {
        M0[r * NP + j] -= f * M0[p * NP + j];
        M1[r * NP + j] -= f * M1[p * NP + j];
      }
    }
    __syncthreads();
  }

  // ---------- phase C: prototypes + label propagation epochs ----------
  if (tid < NP) {           // feature norms (features are now fixed)
    float s = 0.f;
    if (tid < NN) for (int d = 0; d < DIM; ++d) { float v = (float)F[tid * DIM + d]; s += v * v; }
    nrm[tid] = s;
  }
  // proto init: mean over SHOT rows per way (row groups, as in the reference reshape)
  for (int i = tid; i < 8 * DIM; i += BLOCK) {
    int w = i / DIM, d = i - w * DIM;
    float s = 0.f;
    if (w < WAYS) {
      for (int s5 = 0; s5 < SHOT; ++s5) s += (float)F[(w * SHOT + s5) * DIM + d];
      s *= (1.f / SHOT);
    }
    protoF[i] = s;
  }
  __syncthreads();

  for (int ep = 0; ep < NEPOCH; ++ep) {
    get_prob(F, protoF, protoT, Z, nrm, pn, ysL, ust, rsA, flg);
    // Z2 = Inv @ Z  (exact f32 WMMA, K=112)
    if (wid < 7) {
      int ti = wid;
      v8f c = vzero8();
      for (int k0 = 0; k0 < NP; k0 += 4)
        c = wmma_f32(frag_a_f32(M1, NP, ti * 16, k0), frag_b_f32(Z, 16, k0, 0), c);
      int n = ln & 15;
#pragma unroll
      for (int e = 0; e < 8; ++e) {
        int i = ti * 16 + e + ((ln >> 4) << 3);
        Z2[i * 16 + n] = c[e];
      }
    }
    __syncthreads();
    sinkhorn(Z2, 0, NN, (float)(NN / WAYS), NS, ysL, ust, rsA, flg);
    if (tid < WAYS) {
      float s = 0.f;
      for (int i = 0; i < NN; ++i) s += Z2[i * 16 + tid];
      zs[tid] = s;
    }
    __syncthreads();
    // proto <- (1-u)*proto + u * (Z2ᵀ F) / colsum(Z2)   (bf16 WMMA, K=112)
    for (int ch = wid; ch < DIM / 16; ch += 8) {
      v8f c = vzero8();
      for (int k0 = 0; k0 < NPK; k0 += 32)
        c = wmma_bf16(frag_a_zT(Z2, k0), frag_b_F_guard(F, k0, ch * 16), c);
      int n = ln & 15;
#pragma unroll
      for (int e = 0; e < 8; ++e) {
        int w = e + ((ln >> 4) << 3);
        if (w < WAYS) {
          int d = ch * 16 + n;
          protoF[w * DIM + d] = (1.f - UPD) * protoF[w * DIM + d] + UPD * (c[e] / zs[w]);
        }
      }
    }
    __syncthreads();
  }

  // ---------- final probabilities + per-episode query accuracy ----------
  get_prob(F, protoF, protoT, Z, nrm, pn, ysL, ust, rsA, flg);
  if (tid >= NS && tid < NN) {
    float best = -1e30f; int bi = 0;
#pragma unroll
    for (int w = 0; w < WAYS; ++w) { float v = Z[tid * 16 + w]; if (v > best) { best = v; bi = w; } }
    int lbl = yq[b * QS + (tid - NS)];
    rsA[tid] = (bi == lbl) ? 1.f : 0.f;
  }
  __syncthreads();
  if (tid == 0) {
    float s = 0.f;
    for (int i = NS; i < NN; ++i) s += rsA[i];
    out[b] = s / (float)QS;
  }
}

// ---------------- host entry ----------------
extern "C" void kernel_launch(void* const* d_in, const int* in_sizes, int n_in,
                              void* d_out, int out_size, void* d_ws, size_t ws_size,
                              hipStream_t stream) {
  const float* xs = (const float*)d_in[0];
  const float* xq = (const float*)d_in[1];
  const int*   ys = (const int*)d_in[2];
  const int*   yq = (const int*)d_in[3];
  float* out = (float*)d_out;
  bf16*  F   = (bf16*)d_ws;   // needs RUNS*NP*DIM*2 = ~68.4 MB of scratch

  (void)in_sizes; (void)n_in; (void)out_size; (void)ws_size;
  (void)hipFuncSetAttribute((const void*)slp_kernel,
                            hipFuncAttributeMaxDynamicSharedMemorySize, SMEM_BYTES);
  slp_kernel<<<RUNS, BLOCK, SMEM_BYTES, stream>>>(xs, xq, ys, yq, F, out);
}
